// GNN_44727789420934
// MI455X (gfx1250) — matrix operand order repaired and checked by reading, not compiled, not run on hardware
//
#include <hip/hip_runtime.h>
#include <hip/hip_bf16.h>
#include <math.h>

typedef __attribute__((ext_vector_type(16))) _Float16 v16h;
typedef __attribute__((ext_vector_type(8)))  float    v8f;
typedef int i32x4 __attribute__((vector_size(16)));
typedef __attribute__((address_space(1))) i32x4 gl_i32x4;
typedef __attribute__((address_space(3))) i32x4 lds_i32x4;

#define D_HID   128
#define N_TASKS 12

#if defined(__HIP_DEVICE_COMPILE__) && \
    __has_builtin(__builtin_amdgcn_global_load_async_to_lds_b128) && \
    __has_builtin(__builtin_amdgcn_s_wait_asynccnt)
#define GNN_ASYNC_LDS 1
#else
#define GNN_ASYNC_LDS 0
#endif

// ---------------------------------------------------------------------------
// Pack W [128 x 128] f32 row-major (K x N) into a contiguous-k (same as A) WMMA B-operand
// lane layout for V_WMMA_F32_16X16X32_F16, split into f16 hi / f16 lo.
//   lane l: n = l&15, kb = (l>>4)*16, halves j=0..15 -> k = kt*32 + kb + j
// Stored contiguously: idx = ((kt*8+nt)*32 + lane)*16 + j
// ---------------------------------------------------------------------------
__global__ void gnn_pack_w(const float* __restrict__ W,
                           _Float16* __restrict__ whi,
                           _Float16* __restrict__ wlo) {
    int idx = blockIdx.x * blockDim.x + threadIdx.x;      // 0 .. 16383
    if (idx >= 4 * 8 * 32 * 16) return;
    int j    = idx & 15;
    int lane = (idx >> 4) & 31;
    int nt   = (idx >> 9) & 7;
    int kt   = idx >> 12;
    int n  = lane & 15;
    int kb = (lane >> 4) * 16;
    int k  = kt * 32 + kb + j;
    float w = W[(size_t)k * D_HID + nt * 16 + n];
    _Float16 hi = (_Float16)w;
    _Float16 lo = (_Float16)(w - (float)hi);
    whi[idx] = hi;
    wlo[idx] = lo;
}

// ---------------------------------------------------------------------------
// C[M,128] = A[M,128] @ W[128,128] via split-f16 WMMA (fp32-grade accuracy).
// Packed W (64 KB) is staged once per block into LDS via the gfx1250 async
// global->LDS path, then each wave computes a 16-row strip:
// 4 K-steps x 8 N-tiles x 3 WMMA, with B fragments fed by ds_load_b128.
// ---------------------------------------------------------------------------
__global__ void gnn_gemm128(const float* __restrict__ A,
                            const _Float16* __restrict__ bhi,
                            const _Float16* __restrict__ blo,
                            float* __restrict__ C, int mtiles) {
    __shared__ __align__(32) _Float16 sBhi[16384];   // 32 KB
    __shared__ __align__(32) _Float16 sBlo[16384];   // 32 KB

    // ---- stage packed W into LDS (64 KB per block) ----
#if GNN_ASYNC_LDS
    {
        #pragma unroll
        for (int t = 0; t < 8; ++t) {
            int off = (threadIdx.x + t * 256) * 8;   // 16-byte chunks
            __builtin_amdgcn_global_load_async_to_lds_b128(
                (gl_i32x4*)(bhi + off), (lds_i32x4*)(sBhi + off), 0, 0);
            __builtin_amdgcn_global_load_async_to_lds_b128(
                (gl_i32x4*)(blo + off), (lds_i32x4*)(sBlo + off), 0, 0);
        }
        __builtin_amdgcn_s_wait_asynccnt(0);
    }
#else
    {
        const float4* g0 = (const float4*)bhi;
        const float4* g1 = (const float4*)blo;
        float4* s0 = (float4*)sBhi;
        float4* s1 = (float4*)sBlo;
        for (int t = threadIdx.x; t < 2048; t += 256) {
            s0[t] = g0[t];
            s1[t] = g1[t];
        }
    }
#endif
    __syncthreads();

    int wave = threadIdx.x >> 5;
    int lane = threadIdx.x & 31;
    int tile = blockIdx.x * 8 + wave;
    if (tile < mtiles) {                       // wave-uniform: EXEC all-1 for WMMA
        int row0 = tile * 16;
        int m    = lane & 15;
        int hi16 = lane >> 4;                  // 0 or 1
        const float* arow = A + (size_t)(row0 + m) * D_HID;
        int kb_a = hi16 * 16;

        v8f acc[8] = {};

        for (int kt = 0; kt < 4; ++kt) {
            // ---- load + split A fragment (16 f32 -> f16 hi/lo) ----
            const float4* a4 = (const float4*)(arow + kt * 32 + kb_a);
            float4 p0 = a4[0];                 // k = kb..kb+3
            float4 p1 = a4[1];                 // k = kb+4..kb+7
            float4 p2 = a4[2];                 // k = kb+8..kb+11
            float4 p3 = a4[3];                 // k = kb+12..kb+15
            float av[16] = { p0.x, p0.y, p0.z, p0.w,  p1.x, p1.y, p1.z, p1.w,
                             p2.x, p2.y, p2.z, p2.w,  p3.x, p3.y, p3.z, p3.w };
            v16h ah, al;
            #pragma unroll
            for (int j = 0; j < 16; ++j) {
                _Float16 h = (_Float16)av[j];
                ah[j] = h;
                al[j] = (_Float16)(av[j] - (float)h);
            }
            // ---- 8 N-tiles, B fragments from LDS ----
            #pragma unroll
            for (int nt = 0; nt < 8; ++nt) {
                int boff = ((kt * 8 + nt) * 32 + lane) * 16;
                v16h bh = *(const v16h*)(sBhi + boff);
                v16h bl = *(const v16h*)(sBlo + boff);
                acc[nt] = __builtin_amdgcn_wmma_f32_16x16x32_f16(
                              false, ah, false, bh, (short)0, acc[nt], false, false);
                acc[nt] = __builtin_amdgcn_wmma_f32_16x16x32_f16(
                              false, ah, false, bl, (short)0, acc[nt], false, false);
                acc[nt] = __builtin_amdgcn_wmma_f32_16x16x32_f16(
                              false, al, false, bh, (short)0, acc[nt], false, false);
            }
        }

        // ---- store C (f32 16x16 C/D layout: lane l col=l&15, VGPR r -> M=r+hi16*8)
        int n = lane & 15;
        #pragma unroll
        for (int nt = 0; nt < 8; ++nt) {
            #pragma unroll
            for (int r = 0; r < 8; ++r) {
                int mr = r + hi16 * 8;
                C[(size_t)(row0 + mr) * D_HID + nt * 16 + n] = acc[nt][r];
            }
        }
    }
}

// ---------------------------------------------------------------------------
__global__ void gnn_zero(float4* __restrict__ p, int n4) {
    int i = blockIdx.x * blockDim.x + threadIdx.x;
    if (i < n4) p[i] = make_float4(0.f, 0.f, 0.f, 0.f);
}

// Edge scatter: one WAVE = one edge. e = blockIdx*8 + (tid>>5) is wave-uniform
// on wave32, so src/dst become scalar loads; each lane gathers a float4 of the
// source row and issues 4 non-returning global_atomic_add_f32 (L2-resident).
__global__ void gnn_scatter(const float* __restrict__ xw, float* __restrict__ agg,
                            const long long* __restrict__ src,
                            const long long* __restrict__ dst, int n_edges) {
    int wave = threadIdx.x >> 5;
    int lane = threadIdx.x & 31;
    int e = blockIdx.x * 8 + wave;
    if (e >= n_edges) return;
    long long s = src[e];
    long long d = dst[e];
    int c = lane * 4;
    float4 v = *(const float4*)(xw + (size_t)s * D_HID + c);
    float* dp = agg + (size_t)d * D_HID + c;
    atomicAdd(dp + 0, v.x);
    atomicAdd(dp + 1, v.y);
    atomicAdd(dp + 2, v.z);
    atomicAdd(dp + 3, v.w);
}

// h = relu(agg + xw + b)  (in place into agg)
__global__ void gnn_combine_relu(float* __restrict__ agg, const float* __restrict__ xw,
                                 const float* __restrict__ b, int n4) {
    int i = blockIdx.x * blockDim.x + threadIdx.x;
    if (i >= n4) return;
    size_t o = (size_t)i * 4;
    int c = (int)(o & (D_HID - 1));
    float4 a = *(const float4*)(agg + o);
    float4 x = *(const float4*)(xw + o);
    float4 r;
    r.x = a.x + x.x + b[c + 0];
    r.y = a.y + x.y + b[c + 1];
    r.z = a.z + x.z + b[c + 2];
    r.w = a.w + x.w + b[c + 3];
    r.x = r.x > 0.f ? r.x : 0.f;
    r.y = r.y > 0.f ? r.y : 0.f;
    r.z = r.z > 0.f ? r.z : 0.f;
    r.w = r.w > 0.f ? r.w : 0.f;
    *(float4*)(agg + o) = r;
}

// out[n,t] = sigmoid(h[n,:] . Wf[:,t] + bf[t])   (0.3 GFLOP, L2-fed)
__global__ void gnn_head(const float* __restrict__ h, const float* __restrict__ Wf,
                         const float* __restrict__ bf, float* __restrict__ out,
                         int n_nodes) {
    int idx = blockIdx.x * blockDim.x + threadIdx.x;
    if (idx >= n_nodes * N_TASKS) return;
    int node = idx / N_TASKS;
    int t    = idx - node * N_TASKS;
    const float* hr = h + (size_t)node * D_HID;
    float s = bf[t];
    #pragma unroll 8
    for (int k = 0; k < D_HID; ++k) s += hr[k] * Wf[k * N_TASKS + t];
    out[idx] = 1.0f / (1.0f + __expf(-s));
}

__global__ void gnn_scan(const long long* __restrict__ sizes,
                         long long* __restrict__ offs, int g) {
    if (blockIdx.x == 0 && threadIdx.x == 0) {
        long long acc = 0;
        for (int i = 0; i < g; ++i) { offs[i] = acc; acc += sizes[i]; }
        offs[g] = acc;
    }
}

__global__ void gnn_readout(const float* __restrict__ out,
                            const long long* __restrict__ offs,
                            float* __restrict__ res, int g) {
    int idx = blockIdx.x * blockDim.x + threadIdx.x;
    if (idx >= g * N_TASKS) return;
    int gi = idx / N_TASKS;
    int t  = idx - gi * N_TASKS;
    long long a = offs[gi], b = offs[gi + 1];
    float s = 0.f;
    for (long long n = a; n < b; ++n) s += out[n * N_TASKS + t];
    res[idx] = s / (float)(b - a);
}

// ---------------------------------------------------------------------------
extern "C" void kernel_launch(void* const* d_in, const int* in_sizes, int n_in,
                              void* d_out, int out_size, void* d_ws, size_t ws_size,
                              hipStream_t stream) {
    (void)n_in; (void)out_size; (void)ws_size;
    const float*     X    = (const float*)d_in[0];
    const long long* ei   = (const long long*)d_in[1];
    const long long* gs   = (const long long*)d_in[2];
    const float*     W0   = (const float*)d_in[3];
    const float*     b0   = (const float*)d_in[4];
    const float*     W1   = (const float*)d_in[5];
    const float*     b1   = (const float*)d_in[6];
    const float*     Wf   = (const float*)d_in[7];
    const float*     bf   = (const float*)d_in[8];
    float*           res  = (float*)d_out;

    const int N  = in_sizes[0] / D_HID;       // 100000
    const int E  = in_sizes[1] / 2;           // 1600000
    const int G  = in_sizes[2];               // 1000
    const long long* src = ei;
    const long long* dst = ei + E;

    // workspace carve-up (all 256B-aligned for this N)
    char*  ws   = (char*)d_ws;
    size_t NB   = (size_t)N * D_HID * sizeof(float);       // 51.2 MB
    size_t OB   = (size_t)N * N_TASKS * sizeof(float);     // 4.8 MB
    float*     buf_xw  = (float*)(ws);                     // xw / xw2
    float*     buf_agg = (float*)(ws + NB);                // agg -> h (in place)
    float*     buf_out = (float*)(ws + 2 * NB);            // sigmoid outputs
    _Float16*  w0hi = (_Float16*)(ws + 2 * NB + OB);
    _Float16*  w0lo = w0hi + 16384;
    _Float16*  w1hi = w0lo + 16384;
    _Float16*  w1lo = w1hi + 16384;
    long long* offs = (long long*)((char*)(w1lo + 16384));

    const int mtiles   = N / 16;                            // 6250
    const int gemmBlk  = (mtiles + 7) / 8;                  // 8 waves / block
    const int n4       = N * D_HID / 4;
    const int zcBlk    = (n4 + 255) / 256;
    const int scatBlk  = (E + 7) / 8;                       // 1 wave per edge
    const int headBlk  = (N * N_TASKS + 255) / 256;
    const int roBlk    = (G * N_TASKS + 255) / 256;

    // pack weights into WMMA B layout (hi/lo split)
    gnn_pack_w<<<64, 256, 0, stream>>>(W0, w0hi, w0lo);
    gnn_pack_w<<<64, 256, 0, stream>>>(W1, w1hi, w1lo);

    // ---- layer 1 ----
    gnn_gemm128<<<gemmBlk, 256, 0, stream>>>(X, w0hi, w0lo, buf_xw, mtiles);
    gnn_zero<<<zcBlk, 256, 0, stream>>>((float4*)buf_agg, n4);
    gnn_scatter<<<scatBlk, 256, 0, stream>>>(buf_xw, buf_agg, src, dst, E);
    gnn_combine_relu<<<zcBlk, 256, 0, stream>>>(buf_agg, buf_xw, b0, n4);  // buf_agg = h1

    // ---- layer 2 ----
    gnn_gemm128<<<gemmBlk, 256, 0, stream>>>(buf_agg, w1hi, w1lo, buf_xw, mtiles);
    gnn_zero<<<zcBlk, 256, 0, stream>>>((float4*)buf_agg, n4);
    gnn_scatter<<<scatBlk, 256, 0, stream>>>(buf_xw, buf_agg, src, dst, E);
    gnn_combine_relu<<<zcBlk, 256, 0, stream>>>(buf_agg, buf_xw, b1, n4);  // buf_agg = h2

    // ---- head + readout ----
    gnn_head<<<headBlk, 256, 0, stream>>>(buf_agg, Wf, bf, buf_out, N);
    gnn_scan<<<1, 32, 0, stream>>>(gs, offs, G);
    gnn_readout<<<roBlk, 256, 0, stream>>>(buf_out, offs, res, G);
}
